// MultiHeadAttention_23819888624347
// MI455X (gfx1250) — compile-verified, matrix operand order checked by reading, not run on hardware
//
#include <hip/hip_runtime.h>
#include <hip/hip_bf16.h>
#include <math.h>
#include <stdint.h>

// MHA forward for MI455X (gfx1250, wave32, WMMA + TDM).
// B=8, S=1024, D=768, H=12, DK=64.
// d_out = out[8192,768] f32  ++  attn[8,12,1024,1024] f32.
//
// Roofline: ~64 GFLOP vs 403 MB mandatory attn writes -> HBM-write bound.
// GEMMs: 128x128 block tiles, A/B staged in LDS by tensor_load_to_lds
// (double buffered, TENSORcnt), 8 waves x 8 v_wmma_f32_16x16x32_bf16 per
// k-step. TDM pad (16 dwords -> +4 dwords) gives 80B LDS row stride:
// bank-conflict-free ds_load_b128 fragments.

#define Bq 8
#define Sq 1024
#define Dq 768
#define Hq 12
#define DKq 64

typedef __bf16 bf16_t;
typedef float        v8f   __attribute__((ext_vector_type(8)));
typedef bf16_t       v8bf  __attribute__((ext_vector_type(8)));
typedef bf16_t       v16bf __attribute__((ext_vector_type(16)));
typedef unsigned int u32x4 __attribute__((ext_vector_type(4)));
typedef int          i32x8 __attribute__((ext_vector_type(8)));
typedef int          i32x4 __attribute__((ext_vector_type(4)));

union Frag { v16bf v; v8bf h[2]; };

// ---- bf16 fragment helpers (ISA 7.12.2 16-bit A/B layout) ----------------
// half=0 lanes hold K={0..7,16..23}, half=1 K={8..15,24..31}.
__device__ inline v16bf load_frag(const bf16_t* base, int half) {
    Frag u;
    u.h[0] = *(const v8bf*)(base + half * 8);
    u.h[1] = *(const v8bf*)(base + 16 + half * 8);
    return u.v;
}

#define LDS_ROW_B 80   // 64B tile row + 16B TDM pad
__device__ inline v16bf lds_frag(const char* tile, int row, int half) {
    const char* p = tile + row * LDS_ROW_B;
    Frag u;
    u.h[0] = *(const v8bf*)(p + half * 16);
    u.h[1] = *(const v8bf*)(p + 32 + half * 16);
    return u.v;
}

__device__ inline v8f wmma_bf16(v16bf a, v16bf b, v8f c) {
    return __builtin_amdgcn_wmma_f32_16x16x32_bf16(
        false, a, false, b, (short)0, c, false, false);
}

// ---- TDM: 2-D tile (128 rows x 32 bf16) global -> LDS --------------------
// D# per CDNA5 ISA.  Padding: after every 16
// DWORDs (one 64B tile row) insert 4 DWORDs -> 80B LDS row stride.
__device__ inline void tdm_load_tile(unsigned lds_off, const bf16_t* gptr,
                                     int tensor_rows) {
    unsigned long long ga = (unsigned long long)(uintptr_t)gptr;
    u32x4 g0;
    g0[0] = 1u;                                   // count=1, user desc
    g0[1] = lds_off;                              // lds_addr
    g0[2] = (unsigned)(ga & 0xFFFFFFFFu);         // global_addr[31:0]
    g0[3] = (unsigned)((ga >> 32) & 0x1FFFFFFu)   // global_addr[56:32]
            | (2u << 30);                         // type = 2 (image)
    i32x8 g1;
    g1[0] = (1 << 16)        // data_size = 2 bytes
          | (1 << 20)        // pad_enable
          | (3 << 22)        // pad_interval: 16 DWORDs
          | (3 << 25);       // pad_amount:   4 DWORDs
    g1[1] = (int)(768u << 16);            // tensor_dim0 = 768 (low16<<16)
    g1[2] = (int)((unsigned)tensor_rows << 16);   // tensor_dim1 low16<<16
    g1[3] = (int)(32u << 16);             // tile_dim0 = 32 elements
    g1[4] = 128;                          // tile_dim1 = 128 rows
    g1[5] = 768;                          // tensor_dim0_stride = 768
    g1[6] = 0;
    g1[7] = 0;
    i32x4 z4 = {0, 0, 0, 0};
    i32x8 z8 = {0, 0, 0, 0, 0, 0, 0, 0};
    __builtin_amdgcn_tensor_load_to_lds(g0, g1, z4, z4, z8, 0);
}

// ---------------------------------------------------------------- converts
__global__ void cvt_f32_to_bf16(const float* __restrict__ in,
                                bf16_t* __restrict__ out, int n) {
    int i = blockIdx.x * blockDim.x + threadIdx.x;
    if (i < n) out[i] = (bf16_t)in[i];
}

// W[K,N] f32 -> Wt[N,K] bf16  (K = N = 768)
__global__ void transpose_w_bf16(const float* __restrict__ W,
                                 bf16_t* __restrict__ Wt) {
    int i = blockIdx.x * blockDim.x + threadIdx.x;
    if (i < Dq * Dq) {
        int n = i / Dq;
        int k = i % Dq;
        Wt[i] = (bf16_t)W[(size_t)k * Dq + n];
    }
}

// ---------------------------------------------------------------- GEMM
// C[M,768] = A_bf16[M,768] * Wt_bf16[768,768]^T + bias, M = 8192.
// Block: 128x128 tile, 256 threads (8 waves as 4Mx2N, each 32x64).
// A/B k-slabs (128x32) staged in LDS by TDM, double buffered.
// mode 0: Q  -> [B,H,S,DK] bf16      mode 1: K -> [B,H,S,DK] bf16
// mode 2: Vt -> [B,H,DK,S] bf16      mode 3: out -> [M,768] f32
#define TILE_BYTES (128 * LDS_ROW_B)   // 10240
__global__ __launch_bounds__(256)
void gemm_bf16(const bf16_t* __restrict__ A, const bf16_t* __restrict__ Wt,
               const float* __restrict__ bias, void* __restrict__ dst,
               int mode) {
    __shared__ __align__(16) char smem[4 * TILE_BYTES];   // A0 A1 B0 B1

    const int tid  = threadIdx.x;
    const int lane = tid & 31;
    const int wave = tid >> 5;
    const int l15  = lane & 15, half = lane >> 4;

    const int NTB = Dq / 128;                   // 6 N-blocks
    const int mt  = blockIdx.x / NTB;
    const int nt  = blockIdx.x % NTB;

    const bf16_t* Ag = A  + (size_t)mt * 128 * Dq;   // tile row start
    const bf16_t* Bg = Wt + (size_t)nt * 128 * Dq;

    const unsigned smem_base = (unsigned)(uintptr_t)(void*)smem;

    // prologue: stage k-slab 0 into buffer 0
    if (tid < 32) {
        tdm_load_tile(smem_base + 0 * TILE_BYTES, Ag, Bq * Sq);
        tdm_load_tile(smem_base + 2 * TILE_BYTES, Bg, Dq);
        __builtin_amdgcn_s_wait_tensorcnt(0);
    }
    __syncthreads();

    const int arow0 = (wave >> 1) * 32;          // this wave's 32 M rows
    const int bcol0 = (wave & 1) * 64;           // this wave's 64 N cols
    v8f acc[2][4];
#pragma unroll
    for (int im = 0; im < 2; ++im)
#pragma unroll
        for (int in_ = 0; in_ < 4; ++in_)
            acc[im][in_] = (v8f){0.f,0.f,0.f,0.f,0.f,0.f,0.f,0.f};

    int cur = 0;
    for (int ks = 0; ks < Dq / 32; ++ks) {       // 24 k-steps
        if (tid < 32 && ks + 1 < Dq / 32) {      // prefetch next slab
            int kb = (ks + 1) * 32;
            tdm_load_tile(smem_base + (cur ^ 1) * TILE_BYTES, Ag + kb, Bq * Sq);
            tdm_load_tile(smem_base + (2 + (cur ^ 1)) * TILE_BYTES, Bg + kb, Dq);
        }
        const char* Ab = smem + cur * TILE_BYTES;
        const char* Bb = smem + (2 + cur) * TILE_BYTES;
        v16bf af[2], bf[4];
#pragma unroll
        for (int im = 0; im < 2; ++im)
            af[im] = lds_frag(Ab, arow0 + im * 16 + l15, half);
#pragma unroll
        for (int in_ = 0; in_ < 4; ++in_)
            bf[in_] = lds_frag(Bb, bcol0 + in_ * 16 + l15, half);
#pragma unroll
        for (int im = 0; im < 2; ++im)
#pragma unroll
            for (int in_ = 0; in_ < 4; ++in_)
                acc[im][in_] = wmma_bf16(af[im], bf[in_], acc[im][in_]);

        if (tid < 32) __builtin_amdgcn_s_wait_tensorcnt(0);
        __syncthreads();
        cur ^= 1;
    }

    // epilogue: bias + scatter per mode
    float bv[4];
#pragma unroll
    for (int in_ = 0; in_ < 4; ++in_)
        bv[in_] = bias[nt * 128 + bcol0 + in_ * 16 + l15];

#pragma unroll
    for (int im = 0; im < 2; ++im) {
#pragma unroll
        for (int in_ = 0; in_ < 4; ++in_) {
#pragma unroll
            for (int r = 0; r < 8; ++r) {
                float val = acc[im][in_][r] + bv[in_];
                int mg = mt * 128 + arow0 + im * 16 + r + 8 * half;
                int ng = nt * 128 + bcol0 + in_ * 16 + l15;
                if (mode == 3) {
                    ((float*)dst)[(size_t)mg * Dq + ng] = val;
                } else {
                    int b_ = mg >> 10, s_ = mg & 1023;
                    int h_ = ng >> 6,  d_ = ng & 63;
                    bf16_t* o = (bf16_t*)dst;
                    if (mode == 2)   // Vt[b,h,d,s]
                        o[(((size_t)b_ * Hq + h_) * DKq + d_) * Sq + s_] =
                            (bf16_t)val;
                    else             // Q/K [b,h,s,d]
                        o[(((size_t)b_ * Hq + h_) * Sq + s_) * DKq + d_] =
                            (bf16_t)val;
                }
            }
        }
    }
}

// ---------------------------------------------------------------- attention
// One wave per (b, h, 16-query-row block); scores row-block in 64KB LDS.
__global__ __launch_bounds__(32)
void attn_kernel(const bf16_t* __restrict__ Q, const bf16_t* __restrict__ K,
                 const bf16_t* __restrict__ Vt, float* __restrict__ attn,
                 bf16_t* __restrict__ ctx) {
    __shared__ float sc[16 * Sq];

    int lane = threadIdx.x;
    int l15 = lane & 15, half = lane >> 4;
    int qt = blockIdx.x & 63;
    int h  = (blockIdx.x >> 6) % Hq;
    int b  = blockIdx.x / (64 * Hq);
    size_t bh = (size_t)b * Hq + h;

    int qrow = qt * 16 + l15;
    const bf16_t* qp = Q + (bh * Sq + qrow) * DKq;
    v16bf aq0 = load_frag(qp, half);
    v16bf aq1 = load_frag(qp + 32, half);

    for (int nt = 0; nt < 64; ++nt) {            // scores = Q K^T / 8
        int key = nt * 16 + l15;
        const bf16_t* kp = K + (bh * Sq + key) * DKq;
        v8f s8 = {0.f,0.f,0.f,0.f,0.f,0.f,0.f,0.f};
        s8 = wmma_bf16(aq0, load_frag(kp, half), s8);
        s8 = wmma_bf16(aq1, load_frag(kp + 32, half), s8);
#pragma unroll
        for (int r = 0; r < 8; ++r)
            sc[(r + 8 * half) * Sq + nt * 16 + l15] = s8[r] * 0.125f;
    }
    __syncthreads();

    // softmax: lane pair (l, l+16) splits each row
    float* rp = sc + l15 * Sq + half * 512;
    float mx = -3.0e38f;
    for (int i = 0; i < 512; ++i) mx = fmaxf(mx, rp[i]);
    mx = fmaxf(mx, __shfl_xor(mx, 16, 32));
    float sum = 0.f;
    for (int i = 0; i < 512; ++i) { float e = __expf(rp[i] - mx); rp[i] = e; sum += e; }
    sum += __shfl_xor(sum, 16, 32);
    float inv = 1.f / sum;
    for (int i = 0; i < 512; ++i) rp[i] *= inv;
    __syncthreads();

    // attn out, coalesced (LDS layout == global layout)
    float* apn = attn + (bh * Sq + (size_t)qt * 16) * Sq;
    for (int i = lane; i < 16 * Sq; i += 32) apn[i] = sc[i];

    // P[16,1024] * V[1024,64]
    v8f o0 = {0.f,0.f,0.f,0.f,0.f,0.f,0.f,0.f};
    v8f o1 = o0, o2 = o0, o3 = o0;
    for (int kb = 0; kb < Sq; kb += 32) {
        const float* prow = sc + l15 * Sq + kb;
        v16bf pa;
#pragma unroll
        for (int e = 0; e < 16; ++e) {
            int k = (e < 8) ? (half * 8 + e) : (16 + half * 8 + (e - 8));
            pa[e] = (bf16_t)prow[k];
        }
        const bf16_t* vp = Vt + (bh * DKq + l15) * Sq + kb;
        o0 = wmma_bf16(pa, load_frag(vp,           half), o0);
        o1 = wmma_bf16(pa, load_frag(vp + 16 * Sq, half), o1);
        o2 = wmma_bf16(pa, load_frag(vp + 32 * Sq, half), o2);
        o3 = wmma_bf16(pa, load_frag(vp + 48 * Sq, half), o3);
    }
#pragma unroll
    for (int r = 0; r < 8; ++r) {
        int s = qt * 16 + r + 8 * half;
        size_t rowoff = ((size_t)b * Sq + s) * Dq + h * DKq;
        ctx[rowoff + 0  + l15] = (bf16_t)o0[r];
        ctx[rowoff + 16 + l15] = (bf16_t)o1[r];
        ctx[rowoff + 32 + l15] = (bf16_t)o2[r];
        ctx[rowoff + 48 + l15] = (bf16_t)o3[r];
    }
}

// ---------------------------------------------------------------- launch
extern "C" void kernel_launch(void* const* d_in, const int* in_sizes, int n_in,
                              void* d_out, int out_size, void* d_ws, size_t ws_size,
                              hipStream_t stream) {
    const float* x  = (const float*)d_in[0];
    const float* Wq = (const float*)d_in[1];
    const float* bq = (const float*)d_in[2];
    const float* Wk = (const float*)d_in[3];
    const float* bk = (const float*)d_in[4];
    const float* Wv = (const float*)d_in[5];
    const float* bv = (const float*)d_in[6];
    const float* Wo = (const float*)d_in[7];
    const float* bo = (const float*)d_in[8];

    float* outp = (float*)d_out;
    float* attn = (float*)d_out + (size_t)Bq * Sq * Dq;

    const size_t SZ_XBF = (size_t)Bq * Sq * Dq * 2;
    const size_t SZ_WT  = (size_t)Dq * Dq * 2;
    const size_t SZ_QKV = (size_t)Bq * Hq * Sq * DKq * 2;
    char* w = (char*)d_ws;
    bf16_t* xbf = (bf16_t*)w;  w += SZ_XBF;
    bf16_t* wqt = (bf16_t*)w;  w += SZ_WT;
    bf16_t* wkt = (bf16_t*)w;  w += SZ_WT;
    bf16_t* wvt = (bf16_t*)w;  w += SZ_WT;
    bf16_t* wot = (bf16_t*)w;  w += SZ_WT;
    bf16_t* Qb  = (bf16_t*)w;  w += SZ_QKV;
    bf16_t* Kb  = (bf16_t*)w;  w += SZ_QKV;
    bf16_t* Vtb = (bf16_t*)w;  w += SZ_QKV;
    bf16_t* ctx = (bf16_t*)w;  w += SZ_XBF;

    {
        int n = Bq * Sq * Dq;
        cvt_f32_to_bf16<<<(n + 255) / 256, 256, 0, stream>>>(x, xbf, n);
        int nw = Dq * Dq;
        dim3 g((nw + 255) / 256);
        transpose_w_bf16<<<g, 256, 0, stream>>>(Wq, wqt);
        transpose_w_bf16<<<g, 256, 0, stream>>>(Wk, wkt);
        transpose_w_bf16<<<g, 256, 0, stream>>>(Wv, wvt);
        transpose_w_bf16<<<g, 256, 0, stream>>>(Wo, wot);
    }

    const int blocks = (Bq * Sq / 128) * (Dq / 128);   // 64 * 6 = 384
    gemm_bf16<<<blocks, 256, 0, stream>>>(xbf, wqt, bq, (void*)Qb, 0);
    gemm_bf16<<<blocks, 256, 0, stream>>>(xbf, wkt, bk, (void*)Kb, 1);
    gemm_bf16<<<blocks, 256, 0, stream>>>(xbf, wvt, bv, (void*)Vtb, 2);

    attn_kernel<<<Bq * Hq * (Sq / 16), 32, 0, stream>>>(Qb, Kb, Vtb, attn, ctx);

    gemm_bf16<<<blocks, 256, 0, stream>>>(ctx, wot, bo, (void*)outp, 3);
}
